// CausalGatedLinearAttentionV10_76046690943340
// MI455X (gfx1250) — matrix operand order repaired and checked by reading, not compiled, not run on hardware
//
#include <hip/hip_runtime.h>
#include <hip/hip_bf16.h>

// ---------------------------------------------------------------------------
// CausalGatedLinearAttention for MI455X (gfx1250, wave32, WMMA bf16)
//   B=2, T=2048, D=1024, H=16, dh=64  -> M = B*T = 4096 rows
// ---------------------------------------------------------------------------

typedef __bf16 bf16;
typedef __attribute__((ext_vector_type(8)))  __bf16 v8bf;
typedef __attribute__((ext_vector_type(16))) __bf16 v16bf;
typedef __attribute__((ext_vector_type(8)))  float  v8f;

#define D_MODEL 1024
#define N_HEADS 16
#define D_HEAD  64
#define SEQ_T   2048
#define M_ROWS  4096   // B*T
#define CHUNK   16     // attention chunk length

#define WMMA_BF16(a, b, c) \
    __builtin_amdgcn_wmma_f32_16x16x32_bf16(false, (a), false, (b), (short)0, (c), false, false)

// ---------------------------------------------------------------------------
// WMMA fragment loaders (bf16, 16x16x32). Layouts per CDNA5 ISA 7.12.2:
//  A 16x32: lane<16 -> M=lane,    K halves {kb..kb+7, kb+16..kb+23}, kb=0
//           lane>=16-> M=lane-16, kb=8
//  B 32x16: lane<16 -> N=lane, K {0..7,16..23}; lane>=16 -> N=lane-16, K {8..15,24..31}.
// Works for global or LDS pointers (row-major [M][K] / [N][K]).
// ---------------------------------------------------------------------------
__device__ __forceinline__ v16bf load_a_frag(const bf16* A, int ld, int k0, int lane) {
    int r  = lane & 15;
    int kb = (lane & 16) ? 8 : 0;
    const bf16* p = A + (size_t)r * ld + k0 + kb;
    union { v16bf v; v8bf h[2]; } u;
    u.h[0] = *(const v8bf*)(p);
    u.h[1] = *(const v8bf*)(p + 16);
    return u.v;
}

__device__ __forceinline__ v16bf load_b_frag(const bf16* Bt, int ld, int k0, int lane) {
    int n  = lane & 15;
    int kb = (lane & 16) ? 8 : 0;
    const bf16* p = Bt + (size_t)n * ld + k0 + kb;
    union { v16bf v; v8bf h[2]; } u;
    u.h[0] = *(const v8bf*)(p);
    u.h[1] = *(const v8bf*)(p + 16);
    return u.v;
}

// ---------------------------------------------------------------------------
// Weight convert + transpose:  W fp32 [K][N]  ->  Wt bf16 [N][K]
// ---------------------------------------------------------------------------
__global__ __launch_bounds__(256) void cvt_transpose_kernel(const float* __restrict__ W,
                                                            bf16* __restrict__ Wt,
                                                            int K, int N) {
    size_t idx = (size_t)blockIdx.x * blockDim.x + threadIdx.x;
    size_t total = (size_t)K * N;
    if (idx >= total) return;
    int n = (int)(idx / K);
    int k = (int)(idx % K);
    Wt[idx] = (bf16)W[(size_t)k * N + n];
}

// ---------------------------------------------------------------------------
// LayerNorm: writes bf16 x (for gate GEMM) and bf16 x_norm (for qkv GEMM)
// ---------------------------------------------------------------------------
__global__ __launch_bounds__(256) void ln_kernel(const float* __restrict__ x,
                                                 const float* __restrict__ g,
                                                 const float* __restrict__ be,
                                                 bf16* __restrict__ xbf,
                                                 bf16* __restrict__ xnbf) {
    int row = blockIdx.x;
    int tid = threadIdx.x;
    const float* xr = x + (size_t)row * D_MODEL;

    float v[4], s = 0.f, s2 = 0.f;
#pragma unroll
    for (int i = 0; i < 4; i++) {
        float t = xr[tid + 256 * i];
        v[i] = t; s += t; s2 += t * t;
    }
    __shared__ float r1[256], r2[256];
    r1[tid] = s; r2[tid] = s2;
    __syncthreads();
    for (int o = 128; o > 0; o >>= 1) {
        if (tid < o) { r1[tid] += r1[tid + o]; r2[tid] += r2[tid + o]; }
        __syncthreads();
    }
    float mean = r1[0] * (1.f / D_MODEL);
    float var  = r2[0] * (1.f / D_MODEL) - mean * mean;
    float inv  = rsqrtf(var + 1e-5f);
#pragma unroll
    for (int i = 0; i < 4; i++) {
        int c = tid + 256 * i;
        xbf [(size_t)row * D_MODEL + c] = (bf16)v[i];
        xnbf[(size_t)row * D_MODEL + c] = (bf16)((v[i] - mean) * inv * g[c] + be[c]);
    }
}

// ---------------------------------------------------------------------------
// GEMM core: wave tile 32(M) x 64(N), K-step 32, 8 WMMA per step.
// 8 waves/block arranged 4(M) x 2(N) -> block tile 128x128.
// ---------------------------------------------------------------------------
__device__ __forceinline__ void gemm_core(const bf16* __restrict__ Ab,   // tile row start
                                          const bf16* __restrict__ Bt,   // [N][K]
                                          int col0, int K, int lane,
                                          v8f acc[2][4]) {
    for (int k0 = 0; k0 < K; k0 += 32) {
        v16bf a0 = load_a_frag(Ab, K, k0, lane);
        v16bf a1 = load_a_frag(Ab + (size_t)16 * K, K, k0, lane);
        // feed gfx1250 prefetch path: next K-blocks of this lane's A row
        __builtin_prefetch((const void*)(Ab + (size_t)(lane & 15) * K + k0 + 256), 0, 1);
#pragma unroll
        for (int j = 0; j < 4; j++) {
            v16bf b = load_b_frag(Bt + (size_t)(col0 + 16 * j) * K, K, k0, lane);
            acc[0][j] = WMMA_BF16(a0, b, acc[0][j]);
            acc[1][j] = WMMA_BF16(a1, b, acc[1][j]);
        }
    }
}

// Generic WMMA GEMM: C[M][N] = A*Bt^T + bias.  grid = (N/128, M/128)
__global__ __launch_bounds__(256) void gemm_bias_kernel(const bf16* __restrict__ A,
                                                        const bf16* __restrict__ Bt,
                                                        const float* __restrict__ bias,
                                                        float* __restrict__ C,
                                                        int M, int N, int K) {
    int lane = threadIdx.x & 31;
    int wave = threadIdx.x >> 5;
    int wm = wave >> 1, wn = wave & 1;
    int row0 = blockIdx.y * 128 + wm * 32;
    int col0 = blockIdx.x * 128 + wn * 64;

    v8f acc[2][4] = {};
    gemm_core(A + (size_t)row0 * K, Bt, col0, K, lane, acc);

    int n_l   = lane & 15;
    int m_off = (lane >> 4) * 8;
#pragma unroll
    for (int i = 0; i < 2; i++) {
#pragma unroll
        for (int j = 0; j < 4; j++) {
            int col = col0 + 16 * j + n_l;
            float bb = bias[col];
#pragma unroll
            for (int r = 0; r < 8; r++) {
                int row = row0 + 16 * i + m_off + r;
                C[(size_t)row * N + col] = acc[i][j][r] + bb;
            }
        }
    }
}

// ---------------------------------------------------------------------------
// Gate normalize (in place): g = sigmoid(g); g /= (mean(g) + 1e-5)
// ---------------------------------------------------------------------------
__global__ __launch_bounds__(256) void gate_norm_kernel(float* __restrict__ gate) {
    int row = blockIdx.x;
    int tid = threadIdx.x;
    float* gr = gate + (size_t)row * D_MODEL;
    float v[4], s = 0.f;
#pragma unroll
    for (int i = 0; i < 4; i++) {
        float g = 1.f / (1.f + expf(-gr[tid + 256 * i]));
        v[i] = g; s += g;
    }
    __shared__ float red[256];
    red[tid] = s;
    __syncthreads();
    for (int o = 128; o > 0; o >>= 1) {
        if (tid < o) red[tid] += red[tid + o];
        __syncthreads();
    }
    float inv = 1.f / (red[0] * (1.f / D_MODEL) + 1e-5f);
#pragma unroll
    for (int i = 0; i < 4; i++) gr[tid + 256 * i] = v[i] * inv;
}

// ---------------------------------------------------------------------------
// QKV WMMA GEMM + fused epilogue (bias, gate on q/k, elu+1, head-major bf16)
// grid = (3072/128, 4096/128)
// ---------------------------------------------------------------------------
__global__ __launch_bounds__(256) void gemm_qkv_kernel(const bf16* __restrict__ A,
                                                       const bf16* __restrict__ Bt,
                                                       const float* __restrict__ bias,
                                                       const float* __restrict__ gate,
                                                       bf16* __restrict__ qh,
                                                       bf16* __restrict__ kh,
                                                       bf16* __restrict__ vh) {
    const int K = D_MODEL;
    int lane = threadIdx.x & 31;
    int wave = threadIdx.x >> 5;
    int wm = wave >> 1, wn = wave & 1;
    int row0 = blockIdx.y * 128 + wm * 32;
    int col0 = blockIdx.x * 128 + wn * 64;

    v8f acc[2][4] = {};
    gemm_core(A + (size_t)row0 * K, Bt, col0, K, lane, acc);

    int n_l   = lane & 15;
    int m_off = (lane >> 4) * 8;
#pragma unroll
    for (int i = 0; i < 2; i++) {
#pragma unroll
        for (int j = 0; j < 4; j++) {
            int col   = col0 + 16 * j + n_l;
            int which = col >> 10;            // 0:q 1:k 2:v
            int dcol  = col & 1023;
            int hh    = dcol >> 6;
            int off   = dcol & 63;
            float bb  = bias[col];
#pragma unroll
            for (int r = 0; r < 8; r++) {
                int row = row0 + 16 * i + m_off + r;
                float c = acc[i][j][r] + bb;
                if (which < 2) {
                    c *= gate[(size_t)row * D_MODEL + dcol];
                    c = (c > 0.f ? c : expf(c) - 1.f) + 1.f;   // elu + 1
                }
                int b = row >> 11;
                int t = row & 2047;
                size_t di = (((size_t)(b * N_HEADS + hh) * SEQ_T) + t) * D_HEAD + off;
                bf16 cb = (bf16)c;
                if (which == 0)      qh[di] = cb;
                else if (which == 1) kh[di] = cb;
                else                 vh[di] = cb;
            }
        }
    }
}

// ---------------------------------------------------------------------------
// Chunked causal linear attention, WMMA version. One block (4 waves = 128
// threads) per (b,h). State S[64(d)][64(m)] fp32 persists in accumulator
// registers: wave w owns d in [16w,16w+16), acc[j] covers m in [16j,16j+16).
// Per chunk (C=16):
//   O_inter = Q_c @ S            (bf16 S snapshot via LDS, K=64 -> 2 WMMA/wave)
//   Aqk     = Q_c @ K_c^T        (K=64 -> 2 WMMA, causally masked)
//   O_intra = mask(Aqk) @ V_c    (K=16 padded -> 1 WMMA/wave)
//   S      += K_c^T @ V_c        (K=16 padded -> 4 WMMA/wave)
//   den[t]  = q_t . kcum_prev + rowsum(mask(Aqk))[t]
// ---------------------------------------------------------------------------
__global__ __launch_bounds__(128) void attn_chunk_kernel(const bf16* __restrict__ qh,
                                                         const bf16* __restrict__ kh,
                                                         const bf16* __restrict__ vh,
                                                         bf16* __restrict__ attn_out) {
    int bh = blockIdx.x;                 // 0..31
    int b  = bh >> 4, h = bh & 15;
    int tid  = threadIdx.x;              // 0..127
    int lane = tid & 31;
    int wave = tid >> 5;                 // 0..3
    int n_l   = lane & 15;
    int m_off = (lane >> 4) * 8;

    __shared__ __align__(16) bf16 Qc[16][64];     // [t][d]
    __shared__ __align__(16) bf16 Kc[16][64];     // [t'][d]   (B operand of Q@K^T)
    __shared__ __align__(16) bf16 KcT[64][32];    // [d][t']   k-padded to 32
    __shared__ __align__(16) bf16 VcT[64][32];    // [m][t']   k-padded to 32
    __shared__ __align__(16) bf16 SbfT[64][64];   // [m][d]    bf16 snapshot of S
    __shared__ __align__(16) bf16 Abuf[16][32];   // [t][t']   masked A, k-padded
    __shared__ float kcum[64];
    __shared__ float den_base[16], den_intra[16], den_tot[16];

    // zero pads + kcum once
    for (int idx = tid; idx < 64 * 16; idx += 128) {
        KcT[idx >> 4][16 + (idx & 15)] = (bf16)0.f;
        VcT[idx >> 4][16 + (idx & 15)] = (bf16)0.f;
    }
    for (int idx = tid; idx < 16 * 16; idx += 128)
        Abuf[idx >> 4][16 + (idx & 15)] = (bf16)0.f;
    if (tid < 64) kcum[tid] = 0.f;

    v8f acc[4] = {};                     // persistent S tile (d: wave, m: j)

    const bf16* qb = qh + (size_t)bh * SEQ_T * D_HEAD;
    const bf16* kb = kh + (size_t)bh * SEQ_T * D_HEAD;
    const bf16* vb = vh + (size_t)bh * SEQ_T * D_HEAD;

    __syncthreads();

    for (int chunk = 0; chunk < SEQ_T / CHUNK; chunk++) {
        int t0 = chunk * CHUNK;

        // ---- phase 1: load chunk + snapshot state ----
        {
            int row = tid >> 3;          // 0..15
            int c8  = (tid & 7) * 8;     // 0..56
            size_t base = (size_t)(t0 + row) * D_HEAD + c8;
            v8bf qv = *(const v8bf*)(qb + base);
            *(v8bf*)&Qc[row][c8] = qv;
            v8bf kv = *(const v8bf*)(kb + base);
            *(v8bf*)&Kc[row][c8] = kv;
            v8bf vv = *(const v8bf*)(vb + base);
#pragma unroll
            for (int i = 0; i < 8; i++) {
                KcT[c8 + i][row] = kv[i];
                VcT[c8 + i][row] = vv[i];
            }
        }
#pragma unroll
        for (int j = 0; j < 4; j++)
#pragma unroll
            for (int r = 0; r < 8; r++)
                SbfT[16 * j + n_l][16 * wave + m_off + r] = (bf16)acc[j][r];
        __syncthreads();

        // ---- phase 2: O_inter, Aqk, den pieces ----
        v16bf aq0 = load_a_frag(&Qc[0][0], 64, 0,  lane);
        v16bf aq1 = load_a_frag(&Qc[0][0], 64, 32, lane);
        v8f zero = {};
        v16bf bs0 = load_b_frag(&SbfT[16 * wave][0], 64, 0,  lane);
        v16bf bs1 = load_b_frag(&SbfT[16 * wave][0], 64, 32, lane);
        v8f accO = WMMA_BF16(aq0, bs0, zero);
        accO     = WMMA_BF16(aq1, bs1, accO);

        v16bf bk0 = load_b_frag(&Kc[0][0], 64, 0,  lane);
        v16bf bk1 = load_b_frag(&Kc[0][0], 64, 32, lane);
        v8f accA = WMMA_BF16(aq0, bk0, zero);
        accA     = WMMA_BF16(aq1, bk1, accA);

        float mA[8];
#pragma unroll
        for (int r = 0; r < 8; r++) {
            int m = m_off + r;                       // query time in chunk
            mA[r] = (n_l <= m) ? accA[r] : 0.f;      // causal (inclusive)
        }
        if (wave == 0) {                             // rowsum + Abuf store
#pragma unroll
            for (int r = 0; r < 8; r++) {
                float p = mA[r];
                p += __shfl_xor(p, 1, 32);
                p += __shfl_xor(p, 2, 32);
                p += __shfl_xor(p, 4, 32);
                p += __shfl_xor(p, 8, 32);
                if (n_l == 0) den_intra[m_off + r] = p;
                Abuf[m_off + r][n_l] = (bf16)mA[r];
            }
        }
        if (wave == 1) {                             // den_base[t] = q_t . kcum_prev
            int t  = lane >> 1;
            int d0 = (lane & 1) * 32;
            float p = 0.f;
#pragma unroll
            for (int i = 0; i < 32; i++) p += (float)Qc[t][d0 + i] * kcum[d0 + i];
            p += __shfl_xor(p, 1, 32);
            if ((lane & 1) == 0) den_base[t] = p;
        }
        __syncthreads();

        // ---- phase 3: O_intra, state update, kcum, den_tot ----
        v16bf aA = load_a_frag(&Abuf[0][0],        32, 0, lane);
        v16bf aK = load_a_frag(&KcT[16 * wave][0], 32, 0, lane);
#pragma unroll
        for (int j = 0; j < 4; j++) {
            v16bf bV = load_b_frag(&VcT[16 * j][0], 32, 0, lane);
            if (j == wave) accO = WMMA_BF16(aA, bV, accO);
            acc[j] = WMMA_BF16(aK, bV, acc[j]);
        }
        if (tid < 64) {
            float s = 0.f;
#pragma unroll
            for (int i = 0; i < 16; i++) s += (float)Kc[i][tid];
            kcum[tid] += s;
        }
        if (tid < 16) den_tot[tid] = den_base[tid] + den_intra[tid] + 1e-5f;
        __syncthreads();

        // ---- phase 4: epilogue store ----
        int colg = h * D_HEAD + 16 * wave + n_l;
#pragma unroll
        for (int r = 0; r < 8; r++) {
            int t = m_off + r;
            float o = accO[r] / den_tot[t];
            attn_out[((size_t)(b * SEQ_T + t0 + t)) * D_MODEL + colg] = (bf16)o;
        }
        __syncthreads();
    }
}

// ---------------------------------------------------------------------------
// Host-side launch
// ---------------------------------------------------------------------------
extern "C" void kernel_launch(void* const* d_in, const int* in_sizes, int n_in,
                              void* d_out, int out_size, void* d_ws, size_t ws_size,
                              hipStream_t stream) {
    const float* x      = (const float*)d_in[0];
    const float* ln_g   = (const float*)d_in[1];
    const float* ln_b   = (const float*)d_in[2];
    const float* w_qkv  = (const float*)d_in[3];
    const float* b_qkv  = (const float*)d_in[4];
    const float* w_gate = (const float*)d_in[5];
    const float* b_gate = (const float*)d_in[6];
    const float* w_proj = (const float*)d_in[7];
    const float* b_proj = (const float*)d_in[8];
    float* out = (float*)d_out;

    const int Mr = M_ROWS, D = D_MODEL, N3 = 3 * D_MODEL;

    size_t off = 0;
    auto carve = [&](size_t bytes) -> void* {
        void* p = (char*)d_ws + off;
        off += (bytes + 255) & ~(size_t)255;
        return p;
    };
    bf16* xbf    = (bf16*)carve((size_t)Mr * D * sizeof(bf16));
    bf16* xnbf   = (bf16*)carve((size_t)Mr * D * sizeof(bf16));
    bf16* wtqkv  = (bf16*)carve((size_t)D * N3 * sizeof(bf16));   // [3072][1024]
    bf16* wtgate = (bf16*)carve((size_t)D * D * sizeof(bf16));    // [1024][1024]
    bf16* wtproj = (bf16*)carve((size_t)D * D * sizeof(bf16));
    float* gate  = (float*)carve((size_t)Mr * D * sizeof(float));
    bf16* qh     = (bf16*)carve((size_t)Mr * D * sizeof(bf16));
    bf16* kh     = (bf16*)carve((size_t)Mr * D * sizeof(bf16));
    bf16* vh     = (bf16*)carve((size_t)Mr * D * sizeof(bf16));
    bf16* attnb  = (bf16*)carve((size_t)Mr * D * sizeof(bf16));
    (void)ws_size; (void)in_sizes; (void)n_in; (void)out_size;

    // 1) weight convert+transpose to bf16 [N][K]
    {
        size_t tq = (size_t)D * N3;
        cvt_transpose_kernel<<<(tq + 255) / 256, 256, 0, stream>>>(w_qkv, wtqkv, D, N3);
        size_t td = (size_t)D * D;
        cvt_transpose_kernel<<<(td + 255) / 256, 256, 0, stream>>>(w_gate, wtgate, D, D);
        cvt_transpose_kernel<<<(td + 255) / 256, 256, 0, stream>>>(w_proj, wtproj, D, D);
    }

    // 2) layernorm -> bf16 x, bf16 x_norm
    ln_kernel<<<Mr, 256, 0, stream>>>(x, ln_g, ln_b, xbf, xnbf);

    // 3) gate GEMM (raw x) -> fp32 logits, then sigmoid + mean-normalize
    {
        dim3 grid(D / 128, Mr / 128);
        gemm_bias_kernel<<<grid, 256, 0, stream>>>(xbf, wtgate, b_gate, gate, Mr, D, D);
    }
    gate_norm_kernel<<<Mr, 256, 0, stream>>>(gate);

    // 4) qkv GEMM (x_norm) with fused gate/elu epilogue -> head-major bf16
    {
        dim3 grid(N3 / 128, Mr / 128);
        gemm_qkv_kernel<<<grid, 256, 0, stream>>>(xnbf, wtqkv, b_qkv, gate, qh, kh, vh);
    }

    // 5) chunked causal linear attention (WMMA) -> bf16 [B,T,D]
    attn_chunk_kernel<<<2 * N_HEADS, 128, 0, stream>>>(qh, kh, vh, attnb);

    // 6) projection GEMM + bias -> fp32 output
    {
        dim3 grid(D / 128, Mr / 128);
        gemm_bias_kernel<<<grid, 256, 0, stream>>>(attnb, wtproj, b_proj, out, Mr, D, D);
    }
}
